// SelfAttention_71339406786630
// MI455X (gfx1250) — compile-verified, matrix operand order checked by reading, not run on hardware
//
#include <hip/hip_runtime.h>

// ---------------------------------------------------------------------------
// CDNA5 (gfx1250) fused self-attention, bf16 WMMA path + TDM double buffering.
// Pipeline: proj_qkv (WMMA) -> flash attention (TDM -> LDS -> WMMA + online
//           softmax) -> output projection + residual (WMMA).
// ---------------------------------------------------------------------------

typedef __bf16 bf16_t;
typedef __attribute__((ext_vector_type(16))) __bf16 v16bf;
typedef __attribute__((ext_vector_type(8)))  __bf16 v8bf;
typedef __attribute__((ext_vector_type(8)))  float  v8f;
typedef __attribute__((ext_vector_type(4)))  unsigned int v4u;
typedef __attribute__((ext_vector_type(8)))  unsigned int v8u;

#define B_SZ   8
#define C_IN   256
#define C_KEY  128
#define C_VAL  256
#define NPIX   4096   // 64*64

// -------- fragment helpers ------------------------------------------------
// 16-bit A(16x32)/B(32x16) fragment: lane L (row/col = L&15) holds two
// contiguous 8-element K-runs at offsets off and off+16, off = (L&16)?8:0.
__device__ __forceinline__ int frag_k0(int r, int lane) {
    return ((r & 4) << 2) + ((lane & 16) >> 1) + ((r & 3) << 1);
}

__device__ __forceinline__ v8bf load8(const bf16_t* p) {
    return *(const v8bf*)p;
}

__device__ __forceinline__ v16bf make_frag(v8bf lo, v8bf hi) {
    return __builtin_shufflevector(lo, hi, 0, 1, 2, 3, 4, 5, 6, 7,
                                           8, 9, 10, 11, 12, 13, 14, 15);
}

__device__ __forceinline__ v8f wmma_bf16(v16bf a, v16bf b, v8f c) {
    return __builtin_amdgcn_wmma_f32_16x16x32_bf16(
        /*neg_a=*/false, a, /*neg_b=*/false, b,
        /*c_mod=*/(short)0, c, /*reuse_a=*/false, /*reuse_b=*/false);
}

// A fragment from row-major f32 matrix W[row][col], leading dim ld.
__device__ __forceinline__ v16bf load_frag_a_f32(const float* __restrict__ W,
                                                 int ld, int row0, int col0,
                                                 int lane) {
    v16bf a;
    const float* p = W + (size_t)(row0 + (lane & 15)) * ld + col0;
#pragma unroll
    for (int r = 0; r < 8; ++r) {
        int k = frag_k0(r, lane);
        a[2 * r]     = (__bf16)p[k];
        a[2 * r + 1] = (__bf16)p[k + 1];
    }
    return a;
}

// B fragment (K x N) from f32 matrix with element (k,n) = M[k*ldk + n].
__device__ __forceinline__ v16bf load_frag_b_f32(const float* __restrict__ M,
                                                 int ldk, int k0, int n0,
                                                 int lane) {
    v16bf b;
    const float* p = M + (size_t)k0 * ldk + n0 + (lane & 15);
#pragma unroll
    for (int r = 0; r < 8; ++r) {
        int k = frag_k0(r, lane);
        b[2 * r]     = (__bf16)p[(size_t)k * ldk];
        b[2 * r + 1] = (__bf16)p[(size_t)(k + 1) * ldk];
    }
    return b;
}

// -------- Tensor Data Mover: 2D tile (bf16 elements) global -> LDS --------
// Builds the 2-group D# (count=1, type=2 "image", data_size=2B) in SGPRs and
// issues tensor_load_to_lds.  Tracked by TENSORcnt.
__device__ __forceinline__ unsigned lds_addr_of(const void* p) {
    return (unsigned)(size_t)p;  // flat LDS aperture: low 32 bits = LDS offset
}

__device__ __forceinline__ void tdm_load_2d(unsigned lds_byte_addr,
                                            const void* gaddr,
                                            unsigned tensor_d0,
                                            unsigned tensor_d1,
                                            unsigned tile_d0,
                                            unsigned tile_d1,
                                            unsigned long long stride0_elems) {
    unsigned long long ga = (unsigned long long)(size_t)gaddr;
    v4u g0;
    g0[0] = 1u;                                   // count=1, user descriptor
    g0[1] = lds_byte_addr;                        // lds_addr (bytes)
    g0[2] = (unsigned)(ga & 0xffffffffu);         // global_addr[31:0]
    g0[3] = (unsigned)((ga >> 32) & 0x01ffffffu)  // global_addr[56:32]
            | (2u << 30);                         // type = 2 ("image")
    v8u g1;
    g1[0] = (1u << 16);                           // data_size=1 -> 2 bytes
    g1[1] = (tensor_d0 & 0xffffu) << 16;          // tensor_dim0[15:0]
    g1[2] = ((tensor_d0 >> 16) & 0xffffu)         // tensor_dim0[31:16]
            | ((tensor_d1 & 0xffffu) << 16);      // tensor_dim1[15:0]
    g1[3] = ((tensor_d1 >> 16) & 0xffffu)         // tensor_dim1[31:16]
            | ((tile_d0 & 0xffffu) << 16);        // tile_dim0
    g1[4] = (tile_d1 & 0xffffu);                  // tile_dim1 (tile_dim2 = 0)
    g1[5] = (unsigned)(stride0_elems & 0xffffffffu);       // dim0 stride lo
    g1[6] = (unsigned)((stride0_elems >> 32) & 0xffffu);   // dim0 stride hi
    g1[7] = 0u;                                   // dim1 stride (unused, 2D)
    asm volatile("tensor_load_to_lds %0, %1" :: "s"(g0), "s"(g1) : "memory");
}

__device__ __forceinline__ void wait_tensorcnt0() {
#if __has_builtin(__builtin_amdgcn_s_wait_tensorcnt)
    __builtin_amdgcn_s_wait_tensorcnt(0);
#else
    asm volatile("s_wait_tensorcnt 0x0" ::: "memory");
#endif
}

// ---------------------------------------------------------------------------
// Kernel 1: Q/K/V projections.  One wave = one 16(outch) x 16(pixel) tile.
// Q,K stored [B][N][Ck] bf16; V stored [B][Cv][N] bf16.
// ---------------------------------------------------------------------------
__global__ void __launch_bounds__(128) proj_qkv_kernel(
    const float* __restrict__ x,
    const float* __restrict__ Wq, const float* __restrict__ bq,
    const float* __restrict__ Wk, const float* __restrict__ bk,
    const float* __restrict__ Wv, const float* __restrict__ bv,
    bf16_t* __restrict__ Qb, bf16_t* __restrict__ Kb, bf16_t* __restrict__ Vb) {
    const int lane = threadIdx.x & 31;
    const int wid  = blockIdx.x * 4 + (threadIdx.x >> 5);
    const int oct  = wid & 31;          // 0..7 Q, 8..15 K, 16..31 V
    const int nt   = (wid >> 5) & 255;  // pixel tile
    const int b    = wid >> 13;
    const int n0   = nt * 16;

    const float* W;
    const float* bias;
    int oc0, kind;
    if (oct < 8)        { W = Wq; bias = bq; oc0 = oct * 16;        kind = 0; }
    else if (oct < 16)  { W = Wk; bias = bk; oc0 = (oct - 8) * 16;  kind = 1; }
    else                { W = Wv; bias = bv; oc0 = (oct - 16) * 16; kind = 2; }

    const float* xb = x + (size_t)b * C_IN * NPIX;
    v8f acc = {};
#pragma unroll
    for (int cc = 0; cc < 8; ++cc) {
        v16bf a  = load_frag_a_f32(W, C_IN, oc0, cc * 32, lane);
        v16bf bx = load_frag_b_f32(xb, NPIX, cc * 32, n0, lane);
        acc = wmma_bf16(a, bx, acc);
    }

    const int half8 = (lane & 16) ? 8 : 0;
    const int col   = lane & 15;
#pragma unroll
    for (int r = 0; r < 8; ++r) {
        const int row = oc0 + r + half8;
        const float v = acc[r] + bias[row];
        if (kind == 2) {
            Vb[((size_t)b * C_VAL + row) * NPIX + n0 + col] = (bf16_t)v;
        } else {
            bf16_t* dst = (kind == 0) ? Qb : Kb;
            dst[((size_t)(b * NPIX + n0 + col)) * C_KEY + row] = (bf16_t)v;
        }
    }
}

// ---------------------------------------------------------------------------
// Kernel 2: flash attention.  One wave owns a 16-query tile; the 4 waves of a
// block share double-buffered K/V tiles staged into LDS by the TDM.
// ---------------------------------------------------------------------------
__global__ void __launch_bounds__(128) flash_attn_kernel(
    const bf16_t* __restrict__ Qb, const bf16_t* __restrict__ Kb,
    const bf16_t* __restrict__ Vb, bf16_t* __restrict__ Ob) {
    __shared__ __align__(16) bf16_t ldsK[2][32][128];   // 2 x 8 KB
    __shared__ __align__(16) bf16_t ldsV[2][256][32];   // 2 x 16 KB
    __shared__ __align__(16) bf16_t plds[4][16][48];    // per-wave P tile

    const int lane  = threadIdx.x & 31;
    const int wave  = threadIdx.x >> 5;
    const int tile  = blockIdx.x * 4 + wave;  // 0..2047 (block never straddles b)
    const int b     = tile >> 8;
    const int q0    = (tile & 255) << 4;
    const int off   = (lane & 16) ? 8 : 0;
    const int half8 = off;
    const int l15   = lane & 15;

    const bf16_t* Qbase = Qb + (size_t)b * NPIX * C_KEY;
    const bf16_t* Kbase = Kb + (size_t)b * NPIX * C_KEY;
    const bf16_t* Vbase = Vb + (size_t)b * C_VAL * NPIX;

    // Q A-fragments: 4 chunks over Ck=128 (row = query, contiguous channels)
    v16bf qa[4];
    {
        const bf16_t* qp = Qbase + (size_t)(q0 + l15) * C_KEY;
#pragma unroll
        for (int cc = 0; cc < 4; ++cc)
            qa[cc] = make_frag(load8(qp + cc * 32 + off),
                               load8(qp + cc * 32 + 16 + off));
    }

    v8f acc[16] = {};
    float mst[8], lst[8];
#pragma unroll
    for (int r = 0; r < 8; ++r) { mst[r] = -1e30f; lst[r] = 0.0f; }

    const int NT = NPIX / 32;
    // Prologue: TDM-stage tile 0 (K: 32 rows x 128 ch; V: 256 rows x 32 keys)
    if (wave == 0) {
        tdm_load_2d(lds_addr_of(&ldsK[0][0][0]), Kbase,
                    C_KEY, 32, C_KEY, 32, C_KEY);
        tdm_load_2d(lds_addr_of(&ldsV[0][0][0]), Vbase,
                    32, C_VAL, 32, C_VAL, NPIX);
    }

    for (int kt = 0; kt < NT; ++kt) {
        const int buf = kt & 1;
        if (wave == 0) wait_tensorcnt0();   // tile kt resident in ldsK/V[buf]
        __syncthreads();                    // publish to all waves
        if (wave == 0 && (kt + 1) < NT) {   // overlap: stage tile kt+1
            const int kn = (kt + 1) * 32;
            tdm_load_2d(lds_addr_of(&ldsK[buf ^ 1][0][0]),
                        Kbase + (size_t)kn * C_KEY, C_KEY, 32, C_KEY, 32, C_KEY);
            tdm_load_2d(lds_addr_of(&ldsV[buf ^ 1][0][0]),
                        Vbase + kn, 32, C_VAL, 32, C_VAL, NPIX);
        }

        // ---- S = Q * K^T (two 16x16 fragments covering 32 keys) ----
        v8f s0 = {}, s1 = {};
#pragma unroll
        for (int cc = 0; cc < 4; ++cc) {
            const bf16_t* kp0 = &ldsK[buf][l15][cc * 32];
            const bf16_t* kp1 = &ldsK[buf][16 + l15][cc * 32];
            v16bf b0 = make_frag(load8(kp0 + off), load8(kp0 + 16 + off));
            v16bf b1 = make_frag(load8(kp1 + off), load8(kp1 + 16 + off));
            s0 = wmma_bf16(qa[cc], b0, s0);
            s1 = wmma_bf16(qa[cc], b1, s1);
        }

        // ---- online softmax (row = r + half8, cols striped over 16 lanes) --
        float scale[8];
#pragma unroll
        for (int r = 0; r < 8; ++r) {
            float m = fmaxf(s0[r], s1[r]);
            m = fmaxf(m, __shfl_xor(m, 1, 16));
            m = fmaxf(m, __shfl_xor(m, 2, 16));
            m = fmaxf(m, __shfl_xor(m, 4, 16));
            m = fmaxf(m, __shfl_xor(m, 8, 16));
            const float mnew = fmaxf(mst[r], m);
            scale[r] = __expf(mst[r] - mnew);
            mst[r]   = mnew;
        }
#pragma unroll
        for (int r = 0; r < 8; ++r) {
            const float p0 = __expf(s0[r] - mst[r]);
            const float p1 = __expf(s1[r] - mst[r]);
            plds[wave][r + half8][l15]      = (bf16_t)p0;
            plds[wave][r + half8][l15 + 16] = (bf16_t)p1;
            float rs = p0 + p1;
            rs += __shfl_xor(rs, 1, 16);
            rs += __shfl_xor(rs, 2, 16);
            rs += __shfl_xor(rs, 4, 16);
            rs += __shfl_xor(rs, 8, 16);
            lst[r] = lst[r] * scale[r] + rs;
        }

        // ---- rescale O accumulators only when the running max moved ----
        float smin = scale[0];
#pragma unroll
        for (int r = 1; r < 8; ++r) smin = fminf(smin, scale[r]);
        if (__ballot(smin < 0.999999f)) {
#pragma unroll
            for (int f = 0; f < 16; ++f)
#pragma unroll
                for (int r = 0; r < 8; ++r) acc[f][r] *= scale[r];
        }

        __builtin_amdgcn_wave_barrier();  // order LDS P writes vs reads

        // P: C-layout -> A-layout via per-wave LDS tile (DS in-order per wave)
        const bf16_t* pp = &plds[wave][l15][0];
        v16bf pa = make_frag(*(const v8bf*)(pp + off),
                             *(const v8bf*)(pp + 16 + off));

        // ---- O^T += P * V^T : LDS V tile [v][key], keys contiguous ----
#pragma unroll
        for (int f = 0; f < 16; ++f) {
            const bf16_t* vp = &ldsV[buf][f * 16 + l15][0];
            v16bf vb = make_frag(load8(vp + off), load8(vp + 16 + off));
            acc[f] = wmma_bf16(pa, vb, acc[f]);
        }
    }

    // ---- epilogue: normalize by row sums, store O^T as [B][N][Cv] bf16 ----
#pragma unroll
    for (int r = 0; r < 8; ++r) {
        const float inv = 1.0f / lst[r];
#pragma unroll
        for (int f = 0; f < 16; ++f) acc[f][r] *= inv;
    }
    bf16_t* obase = Ob + (size_t)(b * NPIX + q0) * C_VAL;
#pragma unroll
    for (int r = 0; r < 8; ++r)
#pragma unroll
        for (int f = 0; f < 16; ++f)
            obase[(size_t)(r + half8) * C_VAL + f * 16 + l15] = (bf16_t)acc[f][r];
}

// ---------------------------------------------------------------------------
// Kernel 3: out = gamma * (Wr @ O + br) + x.  One wave = 16(c) x 16(n) tile.
// ---------------------------------------------------------------------------
__global__ void __launch_bounds__(128) out_proj_kernel(
    const bf16_t* __restrict__ Ob, const float* __restrict__ Wr,
    const float* __restrict__ br, const float* __restrict__ gamma,
    const float* __restrict__ x, float* __restrict__ out) {
    const int lane = threadIdx.x & 31;
    const int wid  = blockIdx.x * 4 + (threadIdx.x >> 5);
    const int ct   = wid & 15;
    const int nt   = (wid >> 4) & 255;
    const int b    = wid >> 12;
    const int n0   = nt * 16;
    const int c0   = ct * 16;
    const int off  = (lane & 16) ? 8 : 0;

    const bf16_t* Obase = Ob + (size_t)b * NPIX * C_VAL;
    v8f acc = {};
#pragma unroll
    for (int vc = 0; vc < 8; ++vc) {
        v16bf a = load_frag_a_f32(Wr, C_VAL, c0, vc * 32, lane);
        // B fragment: element (k=v, n) = Obase[n*Cv + v], contiguous per lane
        const bf16_t* p = Obase + (size_t)(n0 + (lane & 15)) * C_VAL + vc * 32;
        v16bf bo = make_frag(load8(p + off), load8(p + 16 + off));
        acc = wmma_bf16(a, bo, acc);
    }

    const float g   = gamma[0];
    const int half8 = (lane & 16) ? 8 : 0;
    const int col   = lane & 15;
#pragma unroll
    for (int r = 0; r < 8; ++r) {
        const int c = c0 + r + half8;
        const size_t idx = ((size_t)b * C_IN + c) * NPIX + n0 + col;
        out[idx] = g * (acc[r] + br[c]) + x[idx];
    }
}

// ---------------------------------------------------------------------------
extern "C" void kernel_launch(void* const* d_in, const int* in_sizes, int n_in,
                              void* d_out, int out_size, void* d_ws,
                              size_t ws_size, hipStream_t stream) {
    const float* x     = (const float*)d_in[0];
    const float* Wk    = (const float*)d_in[1];
    const float* bk    = (const float*)d_in[2];
    const float* Wq    = (const float*)d_in[3];
    const float* bq    = (const float*)d_in[4];
    const float* Wv    = (const float*)d_in[5];
    const float* bv    = (const float*)d_in[6];
    const float* Wr    = (const float*)d_in[7];
    const float* br    = (const float*)d_in[8];
    const float* gamma = (const float*)d_in[9];
    float* out = (float*)d_out;

    char* ws = (char*)d_ws;
    bf16_t* Qb = (bf16_t*)(ws);                        //  8 MB  [B][N][Ck]
    bf16_t* Kb = (bf16_t*)(ws + ((size_t)8 << 20));    //  8 MB  [B][N][Ck]
    bf16_t* Vb = (bf16_t*)(ws + ((size_t)16 << 20));   // 16 MB  [B][Cv][N]
    bf16_t* Ob = (bf16_t*)(ws + ((size_t)32 << 20));   // 16 MB  [B][N][Cv]

    // 8 batches * 256 pixel-tiles * 32 outch-tiles = 65536 waves / 4 per block
    proj_qkv_kernel<<<16384, 128, 0, stream>>>(x, Wq, bq, Wk, bk, Wv, bv,
                                               Qb, Kb, Vb);
    // 8 batches * 256 query-tiles = 2048 waves / 4 per block
    flash_attn_kernel<<<512, 128, 0, stream>>>(Qb, Kb, Vb, Ob);
    // 8 batches * 256 pixel-tiles * 16 ch-tiles = 32768 waves / 4 per block
    out_proj_kernel<<<8192, 128, 0, stream>>>(Ob, Wr, br, gamma, x, out);
}
